// RelPositionMultiHeadAttention_83648783057720
// MI455X (gfx1250) — matrix operand order repaired and checked by reading, not compiled, not run on hardware
//
#include <hip/hip_runtime.h>
#include <hip/hip_bf16.h>

// Problem constants (from reference)
#define B_SZ  2
#define S_LEN 2048
#define H_N   8
#define DH    64
#define DM    512
#define P_LEN 4095   // 2*S - 1

typedef __attribute__((ext_vector_type(16))) __bf16 v16bf;
typedef __attribute__((ext_vector_type(8)))  float  v8f;

static __device__ __forceinline__ unsigned short f2bf(float f) {
  union { float f; unsigned u; } x; x.f = f;
  unsigned r = x.u + 0x7FFFu + ((x.u >> 16) & 1u);  // round-to-nearest-even
  return (unsigned short)(r >> 16);
}

static __device__ __forceinline__ v8f wmma_bf16(v16bf a, v16bf b, v8f c) {
  return __builtin_amdgcn_wmma_f32_16x16x32_bf16(false, a, false, b, (short)0, c, false, false);
}

// ---- CDNA5 async copy: global -> LDS, tracked by ASYNCcnt (CDNA5 ISA) ----
// Per-lane: LDS[vdst] = MEM[vaddr], 16 bytes. VDST carries the LDS byte address.
static __device__ __forceinline__ void async_load_b128_to_lds(unsigned lds_addr, const void* gaddr) {
  asm volatile("global_load_async_to_lds_b128 %0, %1, off"
               :: "v"(lds_addr), "v"(gaddr) : "memory");
}
static __device__ __forceinline__ void wait_asynccnt0() {
  asm volatile("s_wait_asynccnt 0" ::: "memory");
}

// ---- WMMA operand loaders: every lane reads contiguous 16B blocks -> *_load_b128 ----

// A-matrix 16x32 bf16 (M = lane%16, h = lane/16): element e -> K = 16h + e
// ==> 32 contiguous bytes ==> two b128 loads.
static __device__ __forceinline__ v16bf load_A_tile(const unsigned short* A, int ld,
                                                    int m0, int k0, int lane, int mclamp) {
  int m = m0 + (lane & 15); if (m > mclamp) m = mclamp;
  int h = lane >> 4;
  const uint4* p = (const uint4*)(A + (size_t)m * ld + k0 + 16 * h);
  union { v16bf v; uint4 q[2]; } r;
  r.q[0] = p[0];
  r.q[1] = p[1];   // +16 bytes = +8 elements (K base 16h+8)
  return r.v;
}

// B-matrix 32x16 where B[K][N] = mem[(row0+N)*ld + k0 + K]; lane N = lane%16, h = lane/16,
// element e -> K = 16h + e  ==> 32 contiguous bytes ==> two b128 loads.
static __device__ __forceinline__ v16bf load_B_t(const unsigned short* base, int ld,
                                                 int row0, int k0, int lane, int rclamp) {
  int row = row0 + (lane & 15); if (row > rclamp) row = rclamp;
  int h = lane >> 4;
  const uint4* p = (const uint4*)(base + (size_t)row * ld + k0 + 16 * h);
  union { v16bf v; uint4 q[2]; } r;
  r.q[0] = p[0];
  r.q[1] = p[1];
  return r.v;
}

// ---- conversions ----
__global__ void cvt_f32_bf16(const float* __restrict__ in, unsigned short* __restrict__ out, int n) {
  int i = blockIdx.x * blockDim.x + threadIdx.x;
  if (i < n) out[i] = f2bf(in[i]);
}

// W (KxN f32, row-major) -> Wt (NxK bf16): makes every GEMM B-load contiguous per lane.
__global__ void cvt_transpose_bf16(const float* __restrict__ in, unsigned short* __restrict__ out,
                                   int K, int N) {
  int k = blockIdx.y * 16 + threadIdx.y;
  int n = blockIdx.x * 16 + threadIdx.x;
  if (k < K && n < N) out[(size_t)n * K + k] = f2bf(in[(size_t)k * N + n]);
}

// ---- Tiled GEMM: C[Mx512] = A[Mx512](bf16) * Wt^T + bias. One wave = 16x64 strip
// (4 N-tiles share each A fetch -> 4 WMMAs per A load).
// mode 0: f32 out; mode 1: bf16 out; mode 2: dual bf16 out (+u, +v); mode 3: bf16 out in
// V-transposed layout vt[((b*H+h)*DH+d)*S + s]  (so attention PV B-loads are contiguous).
__global__ __launch_bounds__(32) void gemm16_bf16(
    const unsigned short* __restrict__ A, const unsigned short* __restrict__ Wt,
    const float* __restrict__ bias, int M, int mode,
    void* out0, void* out1, const float* __restrict__ ub, const float* __restrict__ vb2)
{
  int lane = threadIdx.x;
  int n0 = blockIdx.x * 64, m0 = blockIdx.y * 16;
  v8f acc[4];
#pragma unroll
  for (int nt = 0; nt < 4; ++nt)
#pragma unroll
    for (int i = 0; i < 8; ++i) acc[nt][i] = 0.f;

#pragma unroll 2
  for (int kk = 0; kk < DM; kk += 32) {
    v16bf a = load_A_tile(A, DM, m0, kk, lane, M - 1);
#pragma unroll
    for (int nt = 0; nt < 4; ++nt) {
      v16bf b = load_B_t(Wt, DM, n0 + 16 * nt, kk, lane, DM - 1);
      acc[nt] = wmma_bf16(a, b, acc[nt]);
    }
  }

  int hh = lane >> 4;
#pragma unroll
  for (int nt = 0; nt < 4; ++nt) {
    int nn = n0 + 16 * nt + (lane & 15);
    float bval = bias ? bias[nn] : 0.f;
#pragma unroll
    for (int r = 0; r < 8; ++r) {
      int m = m0 + r + 8 * hh;
      if (m >= M) continue;
      float val = acc[nt][r] + bval;
      size_t idx = (size_t)m * DM + nn;
      if (mode == 0)      ((float*)out0)[idx] = val;
      else if (mode == 1) ((unsigned short*)out0)[idx] = f2bf(val);
      else if (mode == 2) {
        ((unsigned short*)out0)[idx] = f2bf(val + ub[nn]);   // q + pos_bias_u
        ((unsigned short*)out1)[idx] = f2bf(val + vb2[nn]);  // q + pos_bias_v
      } else {  // mode 3: V stored transposed per (b,h): vt[b][h][d][s]
        int bb = m >> 11;            // m / S_LEN
        int s  = m & (S_LEN - 1);
        int h2 = nn >> 6, d = nn & 63;
        ((unsigned short*)out0)[(((size_t)bb * H_N + h2) * DH + d) * S_LEN + s] = f2bf(val);
      }
    }
  }
}

// ---- Fused rel-pos flash attention ----
// One wave per (b, h, 16-query tile); keys streamed in 128-key chunks with online softmax.
// K chunk is staged to LDS via CDNA5 async loads (ASYNCcnt); pos[q,j] = qv[q].p[S-1+j-q]
// handled as a 31-row banded GEMM + LDS shift (band start pb = S-16+j0-q0, always in range).
__global__ __launch_bounds__(32) void relpos_attn(
    const unsigned short* __restrict__ qu, const unsigned short* __restrict__ qv,
    const unsigned short* __restrict__ kb, const unsigned short* __restrict__ vt,
    const unsigned short* __restrict__ pp, unsigned short* __restrict__ ao)
{
  const int lane = threadIdx.x;
  const int laneN = lane & 15, laneH = lane >> 4;
  const int q0 = blockIdx.x * 16;
  const int h  = blockIdx.y;
  const int b  = blockIdx.z;

  __shared__ alignas(16) unsigned short kStage[128][64];  // K chunk (async-staged)
  __shared__ float posBuf[16][32];
  __shared__ float scoreBuf[16][128];
  __shared__ alignas(16) unsigned short probBuf[16][128];
  __shared__ float rowMax[16], rowSum[16], rowFac[16];

  if (lane < 16) { rowMax[lane] = -1e30f; rowSum[lane] = 0.f; }
  __syncthreads();

  const unsigned short* qu_b = qu + ((size_t)b * S_LEN) * DM + h * DH;
  const unsigned short* qv_b = qv + ((size_t)b * S_LEN) * DM + h * DH;
  const unsigned short* k_b  = kb + ((size_t)b * S_LEN) * DM + h * DH;
  const unsigned short* vt_b = vt + (((size_t)b * H_N + h) * DH) * S_LEN;  // [d][s]
  const unsigned short* p_h  = pp + h * DH;

  const unsigned kstage_base = (unsigned)(size_t)(void*)&kStage[0][0];  // LDS byte offset

  v16bf aqu0 = load_A_tile(qu_b, DM, q0, 0,  lane, S_LEN - 1);
  v16bf aqu1 = load_A_tile(qu_b, DM, q0, 32, lane, S_LEN - 1);
  v16bf aqv0 = load_A_tile(qv_b, DM, q0, 0,  lane, S_LEN - 1);
  v16bf aqv1 = load_A_tile(qv_b, DM, q0, 32, lane, S_LEN - 1);

  v8f o[4];
#pragma unroll
  for (int d = 0; d < 4; ++d)
#pragma unroll
    for (int i = 0; i < 8; ++i) o[d][i] = 0.f;

  for (int c0 = 0; c0 < S_LEN; c0 += 128) {
    // ---- async-stage K chunk (128 keys x 64 dims bf16 = 16KB) into LDS ----
    {
      const char* src = (const char*)k_b + (size_t)c0 * (DM * 2);
#pragma unroll
      for (int i = 0; i < 32; ++i) {
        int idx  = i * 32 + lane;        // 1024 x 16B transfers
        int row  = idx >> 3;             // key row within chunk
        int colb = (idx & 7) * 16;       // byte column (128B per row slice)
        async_load_b128_to_lds(kstage_base + idx * 16,
                               src + (size_t)row * (DM * 2) + colb);
      }
    }
    // prefetch next chunk of V while async K copy is in flight
    if (c0 + 128 < S_LEN) {
      __builtin_prefetch(vt_b + (size_t)lane * S_LEN + (c0 + 128), 0, 1);
      __builtin_prefetch(vt_b + (size_t)(lane + 32) * S_LEN + (c0 + 128), 0, 1);
    }
    wait_asynccnt0();

    // ---- scores for this 16x128 chunk ----
    for (int t = 0; t < 8; ++t) {
      int j0 = c0 + 16 * t;
      v8f cs;
#pragma unroll
      for (int i = 0; i < 8; ++i) cs[i] = 0.f;
      cs = wmma_bf16(aqu0, load_B_t(&kStage[0][0], DH, 16 * t, 0,  lane, 127), cs);
      cs = wmma_bf16(aqu1, load_B_t(&kStage[0][0], DH, 16 * t, 32, lane, 127), cs);

      int pb = (S_LEN - 16) + j0 - q0;
#pragma unroll
      for (int nt = 0; nt < 2; ++nt) {
        v8f ps;
#pragma unroll
        for (int i = 0; i < 8; ++i) ps[i] = 0.f;
        ps = wmma_bf16(aqv0, load_B_t(p_h, DM, pb + 16 * nt, 0,  lane, P_LEN - 1), ps);
        ps = wmma_bf16(aqv1, load_B_t(p_h, DM, pb + 16 * nt, 32, lane, P_LEN - 1), ps);
#pragma unroll
        for (int r = 0; r < 8; ++r) posBuf[r + 8 * laneH][16 * nt + laneN] = ps[r];
      }
      __syncthreads();
#pragma unroll
      for (int r = 0; r < 8; ++r) {
        int m = r + 8 * laneH;
        scoreBuf[m][16 * t + laneN] = (cs[r] + posBuf[m][15 + laneN - m]) * 0.125f;
      }
      __syncthreads();
    }

    // ---- online softmax over 16x128 chunk (2 lanes per row, 64 cols each) ----
    {
      int row = lane >> 1, half = lane & 1;
      const float4* s4 = (const float4*)&scoreBuf[row][half * 64];
      float mloc = -1e30f;
#pragma unroll
      for (int i = 0; i < 16; ++i) {
        float4 tv = s4[i];
        mloc = fmaxf(mloc, fmaxf(fmaxf(tv.x, tv.y), fmaxf(tv.z, tv.w)));
      }
      float mchunk = fmaxf(mloc, __shfl_xor(mloc, 1));
      float mold = rowMax[row];
      float mnew = fmaxf(mold, mchunk);
      float fac  = __expf(mold - mnew);
      float sloc = 0.f;
      unsigned* prow = (unsigned*)&probBuf[row][half * 64];
#pragma unroll
      for (int i = 0; i < 16; ++i) {
        float4 tv = s4[i];
        float e0 = __expf(tv.x - mnew), e1 = __expf(tv.y - mnew);
        float e2 = __expf(tv.z - mnew), e3 = __expf(tv.w - mnew);
        sloc += (e0 + e1) + (e2 + e3);
        prow[2 * i + 0] = (unsigned)f2bf(e0) | ((unsigned)f2bf(e1) << 16);
        prow[2 * i + 1] = (unsigned)f2bf(e2) | ((unsigned)f2bf(e3) << 16);
      }
      float stot = sloc + __shfl_xor(sloc, 1);
      if (half == 0) {
        rowMax[row] = mnew;
        rowSum[row] = rowSum[row] * fac + stot;
        rowFac[row] = fac;
      }
    }
    __syncthreads();

    // ---- rescale accumulator rows ----
    float fr[8];
#pragma unroll
    for (int r = 0; r < 8; ++r) fr[r] = rowFac[r + 8 * laneH];
#pragma unroll
    for (int d = 0; d < 4; ++d)
#pragma unroll
      for (int r = 0; r < 8; ++r) o[d][r] *= fr[r];

    // ---- O += P(16x128) * V(128x64); V read from transposed layout -> b128 loads ----
    for (int t8 = 0; t8 < 4; ++t8) {
      v16bf ap = load_A_tile(&probBuf[0][0], 128, 0, 32 * t8, lane, 15);
#pragma unroll
      for (int d = 0; d < 4; ++d) {
        v16bf bvv = load_B_t(vt_b, S_LEN, 16 * d, c0 + 32 * t8, lane, DH - 1);
        o[d] = wmma_bf16(ap, bvv, o[d]);
      }
    }
    __syncthreads();
  }

  // ---- finalize ----
  float sm[8];
#pragma unroll
  for (int r = 0; r < 8; ++r) sm[r] = 1.f / rowSum[r + 8 * laneH];
#pragma unroll
  for (int d = 0; d < 4; ++d)
#pragma unroll
    for (int r = 0; r < 8; ++r) {
      int m = r + 8 * laneH;
      ao[((size_t)b * S_LEN + q0 + m) * DM + h * DH + 16 * d + laneN] = f2bf(o[d][r] * sm[r]);
    }
}

extern "C" void kernel_launch(void* const* d_in, const int* in_sizes, int n_in,
                              void* d_out, int out_size, void* d_ws, size_t ws_size,
                              hipStream_t stream) {
  (void)in_sizes; (void)n_in; (void)out_size; (void)ws_size;

  const float* x  = (const float*)d_in[0];
  const float* pe = (const float*)d_in[1];
  const float* Wq = (const float*)d_in[2];
  const float* bq = (const float*)d_in[3];
  const float* Wk = (const float*)d_in[4];
  const float* bk = (const float*)d_in[5];
  const float* Wv = (const float*)d_in[6];
  const float* bv = (const float*)d_in[7];
  const float* Wp = (const float*)d_in[8];
  const float* Wo = (const float*)d_in[9];
  const float* bo = (const float*)d_in[10];
  const float* pu = (const float*)d_in[11];
  const float* pv = (const float*)d_in[12];

  // workspace layout (bytes); all 16B aligned
  char* ws = (char*)d_ws;
  unsigned short* xb   = (unsigned short*)(ws + 0);          // x bf16       (4096x512)
  unsigned short* peb  = (unsigned short*)(ws + 4194304);    // pos_emb bf16 (4095x512)
  unsigned short* wqt  = (unsigned short*)(ws + 8388608);    // Wq^T bf16 (512x512)
  unsigned short* wkt  = (unsigned short*)(ws + 8912896);
  unsigned short* wvt  = (unsigned short*)(ws + 9437184);
  unsigned short* wpt  = (unsigned short*)(ws + 9961472);
  unsigned short* wot  = (unsigned short*)(ws + 10485760);
  unsigned short* qu   = (unsigned short*)(ws + 11010048);   // (q+u)  bf16
  unsigned short* qv   = (unsigned short*)(ws + 15204352);   // (q+vb) bf16
  unsigned short* kbuf = (unsigned short*)(ws + 19398656);   // k bf16 (row-major)
  unsigned short* vt   = (unsigned short*)(ws + 23592960);   // v bf16 transposed [b][h][d][s]
  unsigned short* pp   = (unsigned short*)(ws + 27787264);   // p bf16 (4095x512)
  unsigned short* ab   = (unsigned short*)(ws + 31981568);   // attn out bf16

  int nx = B_SZ * S_LEN * DM, np = P_LEN * DM;
  cvt_f32_bf16<<<(nx + 255) / 256, 256, 0, stream>>>(x,  xb,  nx);
  cvt_f32_bf16<<<(np + 255) / 256, 256, 0, stream>>>(pe, peb, np);

  dim3 tT(16, 16), gT(DM / 16, DM / 16);
  cvt_transpose_bf16<<<gT, tT, 0, stream>>>(Wq, wqt, DM, DM);
  cvt_transpose_bf16<<<gT, tT, 0, stream>>>(Wk, wkt, DM, DM);
  cvt_transpose_bf16<<<gT, tT, 0, stream>>>(Wv, wvt, DM, DM);
  cvt_transpose_bf16<<<gT, tT, 0, stream>>>(Wp, wpt, DM, DM);
  cvt_transpose_bf16<<<gT, tT, 0, stream>>>(Wo, wot, DM, DM);

  dim3 gProj(DM / 64, (B_SZ * S_LEN) / 16);
  gemm16_bf16<<<gProj, 32, 0, stream>>>(xb, wqt, bq, B_SZ * S_LEN, 2, qu, qv, pu, pv);
  gemm16_bf16<<<gProj, 32, 0, stream>>>(xb, wkt, bk, B_SZ * S_LEN, 1, kbuf, nullptr, nullptr, nullptr);
  gemm16_bf16<<<gProj, 32, 0, stream>>>(xb, wvt, bv, B_SZ * S_LEN, 3, vt, nullptr, nullptr, nullptr);
  dim3 gP(DM / 64, (P_LEN + 15) / 16);
  gemm16_bf16<<<gP, 32, 0, stream>>>(peb, wpt, nullptr, P_LEN, 1, pp, nullptr, nullptr, nullptr);

  dim3 gAttn(S_LEN / 16, H_N, B_SZ);
  relpos_attn<<<gAttn, 32, 0, stream>>>(qu, qv, kbuf, vt, pp, ab);

  gemm16_bf16<<<gProj, 32, 0, stream>>>(ab, wot, bo, B_SZ * S_LEN, 0, d_out, nullptr, nullptr, nullptr);
}
